// GlobalPNAModel_11209864642802
// MI455X (gfx1250) — matrix operand order repaired and checked by reading, not compiled, not run on hardware
//
#include <hip/hip_runtime.h>

typedef __attribute__((ext_vector_type(16))) _Float16 v16h;
typedef __attribute__((ext_vector_type(4)))  _Float16 v4h;
typedef __attribute__((ext_vector_type(8)))  float    v8f;

#define N_NODES   1000000
#define D_FEAT    128
#define N_GRAPHS  8192
#define U_DIM     64
#define HIDDEN    256
#define GDIM      128
#define IN_DIM    576            // U_DIM + 4*D_FEAT, multiple of 32

#define KT1   (IN_DIM / 32)      // 18 K-tiles, GEMM1
#define NT1   (HIDDEN / 16)      // 16 N-tiles, GEMM1
#define KT2   (HIDDEN / 32)      // 8  K-tiles, GEMM2
#define NT2   (GDIM / 16)        // 8  N-tiles, GEMM2
#define W1P_HALVES (KT1 * NT1 * 32 * 16)   // 147,456
#define W2P_HALVES (KT2 * NT2 * 32 * 16)   // 32,768

#define FLT_BIG 3.402823466e38f

// ---------------------------------------------------------------------------
// WMMA fragment helpers (wave32, contiguous-k operand order k = 16g + s)
// ---------------------------------------------------------------------------
union HFrag { v16h v; float4 q[2]; };

// A: 16x32 f16 tile from row-major [M x lda].  lane m = lane&15 picks row,
// group g = lane>>4:  halves 0..15 <- K in [16g,16g+16)
__device__ __forceinline__ v16h load_a_frag(const _Float16* __restrict__ A, int lda,
                                            int mBase, int kb, int lane) {
    int m = lane & 15, g = lane >> 4;
    const _Float16* p = A + (long)(mBase + m) * lda + kb + 16 * g;
    HFrag fr;
    fr.q[0] = *(const float4*)(p);
    fr.q[1] = *(const float4*)(p + 8);
    return fr.v;
}

// B: pre-swizzled fragment — 16 contiguous halves per lane (32B), one b128 pair.
__device__ __forceinline__ v16h load_b_packed(const _Float16* __restrict__ p) {
    HFrag fr;
    fr.q[0] = *(const float4*)(p);
    fr.q[1] = *(const float4*)(p + 8);
    return fr.v;
}

// ---------------------------------------------------------------------------
// Kernel 1: pack weights f32 -> f16 in WMMA-B fragment order.
// Fragment layout: half index i (v=i>>1, p=i&1), lane (n=lane&15, g=lane>>4)
// maps to source element W[(kTile*32 + 16g + 2v + p) * ldb + nTile*16 + n].
// Packed offset: ((kTile*NT + nTile)*32 + lane)*16 + i   (contiguous per lane)
// ---------------------------------------------------------------------------
__global__ void k_pack_weights(const float* __restrict__ W1, const float* __restrict__ W2,
                               _Float16* __restrict__ W1p, _Float16* __restrict__ W2p) {
    int idx = blockIdx.x * blockDim.x + threadIdx.x;
    if (idx < W1P_HALVES) {
        int i    = idx & 15;
        int lane = (idx >> 4) & 31;
        int frag = idx >> 9;
        int nT   = frag % NT1;
        int kT   = frag / NT1;
        int n = lane & 15, g = lane >> 4, v = i >> 1, p = i & 1;
        int K   = kT * 32 + 16 * g + 2 * v + p;
        int col = nT * 16 + n;
        W1p[idx] = (_Float16)W1[K * HIDDEN + col];
    }
    if (idx < W2P_HALVES) {
        int i    = idx & 15;
        int lane = (idx >> 4) & 31;
        int frag = idx >> 9;
        int nT   = frag % NT2;
        int kT   = frag / NT2;
        int n = lane & 15, g = lane >> 4, v = i >> 1, p = i & 1;
        int K   = kT * 32 + 16 * g + 2 * v + p;
        int col = nT * 16 + n;
        W2p[idx] = (_Float16)W2[K * GDIM + col];
    }
}

// ---------------------------------------------------------------------------
// Kernel 2: segment aggregation (batch sorted -> contiguous ranges).
// One block per graph; 128 threads = 4 row-phases x 32 lanes.
// Lane owns features [4*lane, 4*lane+4): one float4 (b128) per row -> a wave
// streams a full 512B row per iteration, 4 waves stream 4 rows (2KB/iter).
// Partials combined across the 4 row-phases via LDS, then the first wave
// finalizes mean/std/max/min and emits the packed f16 A1 row.
// ---------------------------------------------------------------------------
__global__ void __launch_bounds__(128)
k_aggregate(const float* __restrict__ x, const int* __restrict__ batch,
            const float* __restrict__ u, _Float16* __restrict__ A1, int nNodes) {
    int g = blockIdx.x;
    __shared__ int sRange[2];
    __shared__ float4 ps[4][32], ps2[4][32], pmx[4][32], pmn[4][32];   // 8KB

    if (threadIdx.x < 2) {
        int target = g + (int)threadIdx.x;            // lower_bound(batch, target)
        int lo = 0, hi = nNodes;
        while (lo < hi) {
            int mid = (lo + hi) >> 1;
            if (batch[mid] < target) lo = mid + 1; else hi = mid;
        }
        sRange[threadIdx.x] = lo;
    }
    __syncthreads();
    int start = sRange[0], end = sRange[1];
    int lane = threadIdx.x & 31;       // feature group
    int rsub = threadIdx.x >> 5;       // row phase 0..3

    float4 s  = {0.f, 0.f, 0.f, 0.f};
    float4 s2 = {0.f, 0.f, 0.f, 0.f};
    float4 mx = {-FLT_BIG, -FLT_BIG, -FLT_BIG, -FLT_BIG};
    float4 mn = { FLT_BIG,  FLT_BIG,  FLT_BIG,  FLT_BIG};

    for (int n = start + rsub; n < end; n += 4) {
        float4 v = *(const float4*)(x + (long)n * D_FEAT + lane * 4);
        s.x += v.x;  s.y += v.y;  s.z += v.z;  s.w += v.w;
        s2.x += v.x * v.x;  s2.y += v.y * v.y;  s2.z += v.z * v.z;  s2.w += v.w * v.w;
        mx.x = fmaxf(mx.x, v.x);  mx.y = fmaxf(mx.y, v.y);
        mx.z = fmaxf(mx.z, v.z);  mx.w = fmaxf(mx.w, v.w);
        mn.x = fminf(mn.x, v.x);  mn.y = fminf(mn.y, v.y);
        mn.z = fminf(mn.z, v.z);  mn.w = fminf(mn.w, v.w);
    }
    ps[rsub][lane] = s;  ps2[rsub][lane] = s2;
    pmx[rsub][lane] = mx; pmn[rsub][lane] = mn;

    _Float16* row = A1 + (long)g * IN_DIM;
    if (threadIdx.x >= 64) {                       // threads 64..127 copy u (64 vals)
        int f = threadIdx.x - 64;
        row[f] = (_Float16)u[(long)g * U_DIM + f];
    }
    __syncthreads();

    if (threadIdx.x < 32) {
        float S[4]  = {0.f, 0.f, 0.f, 0.f};
        float S2[4] = {0.f, 0.f, 0.f, 0.f};
        float MX[4] = {-FLT_BIG, -FLT_BIG, -FLT_BIG, -FLT_BIG};
        float MN[4] = { FLT_BIG,  FLT_BIG,  FLT_BIG,  FLT_BIG};
#pragma unroll
        for (int r = 0; r < 4; ++r) {
            float4 a = ps[r][lane],  b = ps2[r][lane];
            float4 c = pmx[r][lane], d = pmn[r][lane];
            S[0] += a.x; S[1] += a.y; S[2] += a.z; S[3] += a.w;
            S2[0] += b.x; S2[1] += b.y; S2[2] += b.z; S2[3] += b.w;
            MX[0] = fmaxf(MX[0], c.x); MX[1] = fmaxf(MX[1], c.y);
            MX[2] = fmaxf(MX[2], c.z); MX[3] = fmaxf(MX[3], c.w);
            MN[0] = fminf(MN[0], d.x); MN[1] = fminf(MN[1], d.y);
            MN[2] = fminf(MN[2], d.z); MN[3] = fminf(MN[3], d.w);
        }
        int cnt = end - start;
        float safe = cnt > 0 ? (float)cnt : 1.f;
        v4h vMean, vStd, vMax, vMin;
#pragma unroll
        for (int j = 0; j < 4; ++j) {
            float mean = S[j] / safe;
            float var  = S2[j] / safe - mean * mean;
            float sd   = sqrtf(fmaxf(var, 1e-5f));      // clip(var,1e-5) then sqrt
            float mxv  = cnt > 0 ? MX[j] : 0.f;         // reference zeroes empties
            float mnv  = cnt > 0 ? MN[j] : 0.f;
            vMean[j] = (_Float16)mean;
            vStd[j]  = (_Float16)sd;
            vMax[j]  = (_Float16)mxv;
            vMin[j]  = (_Float16)mnv;
        }
        int f4 = lane * 4;                               // 8B-aligned packed stores
        *(v4h*)(row + U_DIM + 0 * D_FEAT + f4) = vMean;
        *(v4h*)(row + U_DIM + 1 * D_FEAT + f4) = vStd;
        *(v4h*)(row + U_DIM + 2 * D_FEAT + f4) = vMax;
        *(v4h*)(row + U_DIM + 3 * D_FEAT + f4) = vMin;
    }
}

// ---------------------------------------------------------------------------
// Kernel 3: h = LayerNorm(SELU(A1 @ W1 + b1)), fused.
// Block = 16 waves, 16-row x 256-col tile of h; wave w -> columns [16w,16w+16);
// v_wmma_f32_16x16x32_f16 over K=576 (18 steps) with packed-B b128 loads,
// then wave w LayerNorms row w via wave32 shuffles.
// ---------------------------------------------------------------------------
__global__ void __launch_bounds__(512)
k_gemm1_selu_ln(const _Float16* __restrict__ A1, const _Float16* __restrict__ W1p,
                const float* __restrict__ b1, const float* __restrict__ ln_g,
                const float* __restrict__ ln_b, _Float16* __restrict__ hOut) {
    __shared__ float hbuf[16][HIDDEN];
    int lane = threadIdx.x & 31;
    int wave = threadIdx.x >> 5;
    int mBase = blockIdx.x * 16;
    int nBase = wave * 16;

    const _Float16* bp = W1p + ((0 * NT1 + wave) * 32 + lane) * 16;
    const int bStride = NT1 * 32 * 16;                 // halves per K-tile

    v8f c = {};
#pragma unroll
    for (int kt = 0; kt < KT1; ++kt) {
        v16h a = load_a_frag(A1, IN_DIM, mBase, kt * 32, lane);
        v16h b = load_b_packed(bp + kt * bStride);
        c = __builtin_amdgcn_wmma_f32_16x16x32_f16(false, a, false, b,
                                                   (short)0, c, false, false);
    }

    // C/D layout: n = lane&15 (column), m = 8*(lane>>4) + r (row)
    int n = lane & 15, g = lane >> 4;
    float bias = b1[nBase + n];
    const float kScale = 1.0507009873554805f;
    const float kAlpha = 1.6732632423543772f;
#pragma unroll
    for (int r = 0; r < 8; ++r) {
        float v = c[r] + bias;
        v = kScale * (v > 0.f ? v : kAlpha * (__expf(v) - 1.f));   // SELU
        hbuf[8 * g + r][nBase + n] = v;
    }
    __syncthreads();

    // LayerNorm: wave w owns row w (256 values, 8 per lane)
    int row = wave;
    float s = 0.f, s2 = 0.f;
#pragma unroll
    for (int j = 0; j < HIDDEN / 32; ++j) {
        float v = hbuf[row][lane + 32 * j];
        s += v; s2 += v * v;
    }
#pragma unroll
    for (int off = 16; off >= 1; off >>= 1) {
        s  += __shfl_xor(s,  off, 32);
        s2 += __shfl_xor(s2, off, 32);
    }
    float mean = s * (1.f / HIDDEN);
    float var  = s2 * (1.f / HIDDEN) - mean * mean;
    float rstd = rsqrtf(var + 1e-5f);

    _Float16* orow = hOut + (long)(mBase + row) * HIDDEN;
#pragma unroll
    for (int j = 0; j < HIDDEN / 32; ++j) {
        int col = lane + 32 * j;
        float v = (hbuf[row][col] - mean) * rstd * ln_g[col] + ln_b[col];
        orow[col] = (_Float16)v;
    }
}

// ---------------------------------------------------------------------------
// Kernel 4: out = h @ W2 + b2   (M=8192, K=256, N=128); 8 WMMA K-steps.
// ---------------------------------------------------------------------------
__global__ void __launch_bounds__(256)
k_gemm2(const _Float16* __restrict__ H, const _Float16* __restrict__ W2p,
        const float* __restrict__ b2, float* __restrict__ out) {
    int lane = threadIdx.x & 31;
    int wave = threadIdx.x >> 5;
    int mBase = blockIdx.x * 16;
    int nBase = wave * 16;

    const _Float16* bp = W2p + ((0 * NT2 + wave) * 32 + lane) * 16;
    const int bStride = NT2 * 32 * 16;

    v8f c = {};
#pragma unroll
    for (int kt = 0; kt < KT2; ++kt) {
        v16h a = load_a_frag(H, HIDDEN, mBase, kt * 32, lane);
        v16h b = load_b_packed(bp + kt * bStride);
        c = __builtin_amdgcn_wmma_f32_16x16x32_f16(false, a, false, b,
                                                   (short)0, c, false, false);
    }

    int n = lane & 15, g = lane >> 4;
    float bias = b2[nBase + n];
#pragma unroll
    for (int r = 0; r < 8; ++r) {
        out[(long)(mBase + 8 * g + r) * GDIM + nBase + n] = c[r] + bias;
    }
}

// ---------------------------------------------------------------------------
// Launch
// ---------------------------------------------------------------------------
extern "C" void kernel_launch(void* const* d_in, const int* in_sizes, int n_in,
                              void* d_out, int out_size, void* d_ws, size_t ws_size,
                              hipStream_t stream) {
    const float* x     = (const float*)d_in[0];
    // d_in[1] edge_index, d_in[2] edge_attr: unused by the reference result
    const float* u     = (const float*)d_in[3];
    const int*   batch = (const int*)d_in[4];
    const float* W1    = (const float*)d_in[5];
    const float* b1    = (const float*)d_in[6];
    const float* ln_g  = (const float*)d_in[7];
    const float* ln_b  = (const float*)d_in[8];
    const float* W2    = (const float*)d_in[9];
    const float* b2    = (const float*)d_in[10];
    float* out = (float*)d_out;

    // Workspace layout (all 16B aligned)
    char* ws = (char*)d_ws;
    const size_t OFF_A1  = 0;                                         // 9,437,184
    const size_t OFF_W1P = OFF_A1  + (size_t)N_GRAPHS * IN_DIM * 2;
    const size_t OFF_W2P = OFF_W1P + (size_t)W1P_HALVES * 2;
    const size_t OFF_H   = OFF_W2P + (size_t)W2P_HALVES * 2;
    _Float16* A1  = (_Float16*)(ws + OFF_A1);
    _Float16* W1p = (_Float16*)(ws + OFF_W1P);
    _Float16* W2p = (_Float16*)(ws + OFF_W2P);
    _Float16* H   = (_Float16*)(ws + OFF_H);

    // 1. weights -> f16, pre-swizzled into WMMA-B fragment order
    k_pack_weights<<<(W1P_HALVES + 255) / 256, 256, 0, stream>>>(W1, W2, W1p, W2p);
    // 2. segment aggregation + A1 packing (dominant: streams 512MB of x)
    k_aggregate<<<N_GRAPHS, 128, 0, stream>>>(x, batch, u, A1, N_NODES);
    // 3. GEMM1 + SELU + LayerNorm fused (WMMA)
    k_gemm1_selu_ln<<<N_GRAPHS / 16, 512, 0, stream>>>(A1, W1p, b1, ln_g, ln_b, H);
    // 4. GEMM2 + bias (WMMA)
    k_gemm2<<<N_GRAPHS / 16, 256, 0, stream>>>(H, W2p, b2, out);
}
